// GATLayer_75952201663214
// MI455X (gfx1250) — matrix operand order repaired and checked by reading, not compiled, not run on hardware
//
#include <hip/hip_runtime.h>
#include <hip/hip_bf16.h>

typedef __attribute__((ext_vector_type(2)))  float        v2f;
typedef __attribute__((ext_vector_type(8)))  float        v8f;
typedef __attribute__((ext_vector_type(16))) _Float16     v16h;
typedef __attribute__((ext_vector_type(4)))  unsigned int v4u;
typedef __attribute__((ext_vector_type(8)))  unsigned int v8u;

#define GB   16
#define GN   512
#define GF   256
#define GH   8
#define GD   32

// ---------------------------------------------------------------------------
// LDS byte offset of a __shared__ object: the generic-pointer aperture layout
// keeps the wave-relative LDS offset in addr[31:0] (ISA 10.2 aperture table).
// ---------------------------------------------------------------------------
__device__ __forceinline__ unsigned lds_off(const void* p) {
  return (unsigned)(unsigned long long)(uintptr_t)p;
}

// ---------------------------------------------------------------------------
// Tensor Data Mover: build a 2D D# (groups 0+1, VADDR2/3 omitted -> <=2D) and
// issue TENSOR_LOAD_TO_LDS. data_size_code: 0=1B 1=2B 2=4B 3=8B.
// Tile walks Y=0..tile_y-1 rows of tile_x elems, global row stride `stride`
// (elements), packing contiguously into LDS at lds_addr. (ISA ch. 8.2-8.4)
// ---------------------------------------------------------------------------
__device__ __forceinline__ void tdm_load_2d(unsigned lds_addr, const void* gaddr,
                                            unsigned tile_x, unsigned tile_y,
                                            unsigned tensor_dx, unsigned tensor_dy,
                                            unsigned stride, unsigned data_size_code) {
  const unsigned long long ga = (unsigned long long)(uintptr_t)gaddr;
  v4u g0;
  g0[0] = 1u;                                        // count=1, user mode
  g0[1] = lds_addr;                                  // bits 63:32  lds_addr
  g0[2] = (unsigned)(ga & 0xffffffffu);              // bits 95:64  global_addr lo
  g0[3] = (unsigned)((ga >> 32) & 0x01ffffffu)       // bits 120:96 global_addr hi
        | (2u << 30);                                // bits 127:126 type=2 (image)
  v8u g1;
  g1[0] = (data_size_code & 3u) << 16;               // wg_mask=0, data_size
  g1[1] = (tensor_dx & 0xffffu) << 16;               // tensor_dim0[15:0] @ 63:48
  g1[2] = (tensor_dx >> 16) | ((tensor_dy & 0xffffu) << 16);   // dim0 hi | dim1 lo
  g1[3] = (tensor_dy >> 16) | ((tile_x & 0xffffu) << 16);      // dim1 hi | tile_dim0
  g1[4] = (tile_y & 0xffffu);                        // tile_dim1; tile_dim2=0
  g1[5] = stride;                                    // tensor_dim0_stride lo32
  g1[6] = 0u;                                        // stride hi16 | dim1_stride lo
  g1[7] = 0u;
  asm volatile("tensor_load_to_lds %0, %1" :: "s"(g0), "s"(g1) : "memory");
}

// ---------------------------------------------------------------------------
// Kernel 1: Wh = x @ W  (M=8192, N=256, K=256) in full f32 via
// v_wmma_f32_16x16x4_f32. The 16x256 A-tile is DMA'd into LDS with one TDM
// descriptor. 8 waves x 2 n-tiles each. Epilogue writes Wh f32 [B,N,H,D] and
// a transposed f16 copy WhT [B,H,D,N] for the aggregation GEMM's B operand.
// ---------------------------------------------------------------------------
__global__ __launch_bounds__(256)
void gat_gemm_wh(const float* __restrict__ x, const float* __restrict__ W,
                 float* __restrict__ wh32, _Float16* __restrict__ wh16T) {
  __shared__ float lds_a[16 * 256];
  const int m0   = blockIdx.x << 4;
  const int t    = threadIdx.x;
  const int wave = t >> 5;
  const int lane = t & 31;
  const int half = lane >> 4;
  const int nloc = lane & 15;
  const int mloc = lane & 15;

  if (t == 0) {
    tdm_load_2d(lds_off(lds_a), x + (size_t)m0 * GF,
                /*tile*/ 4096, 1, /*tensor*/ 4096, 1, /*stride*/ 4096, /*4B*/ 2);
    __builtin_amdgcn_s_wait_tensorcnt((short)0);
  }
  __syncthreads();

  const int n0a = wave * 16;
  const int n0b = (wave + 8) * 16;
  v8f c0 = {}, c1 = {};

  for (int k = 0; k < 256; k += 4) {
    const int kh = k + half * 2;     // lanes 0-15: K={k,k+1}; 16-31: K={k+2,k+3}
    v2f a, b0, b1;
    a[0] = lds_a[mloc * 256 + kh];
    a[1] = lds_a[mloc * 256 + kh + 1];
    b0[0] = W[(size_t)kh * 256 + n0a + nloc];
    b0[1] = W[(size_t)(kh + 1) * 256 + n0a + nloc];
    b1[0] = W[(size_t)kh * 256 + n0b + nloc];
    b1[1] = W[(size_t)(kh + 1) * 256 + n0b + nloc];
    c0 = __builtin_amdgcn_wmma_f32_16x16x4_f32(false, a, false, b0, (short)0, c0, false, false);
    c1 = __builtin_amdgcn_wmma_f32_16x16x4_f32(false, a, false, b1, (short)0, c1, false, false);
  }

  const int b = m0 >> 9;
#pragma unroll
  for (int r = 0; r < 8; ++r) {
    const int m    = m0 + r + (half << 3);   // C layout: VGPR r -> row r (+8 hi lanes)
    const int node = m & (GN - 1);
    {
      const int col = n0a + nloc;
      const float v = c0[r];
      wh32[(size_t)m * GF + col] = v;
      const int hh = col >> 5, d = col & 31;
      wh16T[((size_t)(b * GH + hh) * GD + d) * GN + node] = (_Float16)v;
    }
    {
      const int col = n0b + nloc;
      const float v = c1[r];
      wh32[(size_t)m * GF + col] = v;
      const int hh = col >> 5, d = col & 31;
      wh16T[((size_t)(b * GH + hh) * GD + d) * GN + node] = (_Float16)v;
    }
  }
}

// ---------------------------------------------------------------------------
// Kernel 2: s1[b,n,h] = Wh[b,n,h,:].a1 ; s2 likewise. One block per node,
// one wave per head, wave-local shuffle reduction.
// ---------------------------------------------------------------------------
__global__ __launch_bounds__(256)
void gat_scores(const float* __restrict__ wh32, const float* __restrict__ a1,
                const float* __restrict__ a2, float* __restrict__ s1,
                float* __restrict__ s2) {
  const int node = blockIdx.x;
  const int t = threadIdx.x;
  const int h = t >> 5, d = t & 31;
  const float v  = wh32[(size_t)node * GF + t];
  float p1 = v * a1[d];
  float p2 = v * a2[d];
#pragma unroll
  for (int o = 16; o > 0; o >>= 1) {
    p1 += __shfl_xor(p1, o, 32);
    p2 += __shfl_xor(p2, o, 32);
  }
  if (d == 0) {
    s1[node * GH + h] = p1;
    s2[node * GH + h] = p2;
  }
}

// ---------------------------------------------------------------------------
// Kernel 3: flash-style attention + aggregation.
// Block = (b, 16-row i-tile); 8 waves = 8 heads. Double-buffered j-tiles:
// wave 0 issues 3 TDM descriptors (adj 16x32, edge_type 16x32, s2 strip) for
// tile n+1, waits tensorcnt<=3 (tile n complete, n+1 in flight), barrier,
// all heads compute scores + online softmax + v_wmma_f32_16x16x32_f16.
// ---------------------------------------------------------------------------
__global__ __launch_bounds__(256)
void gat_attn(const float* __restrict__ s1, const float* __restrict__ s2,
              const _Float16* __restrict__ wh16T, const int* __restrict__ adj,
              const int* __restrict__ et, const float* __restrict__ edge_emb,
              float* __restrict__ out) {
  __shared__ int   lds_adj[2][16 * 32];
  __shared__ int   lds_et [2][16 * 32];
  __shared__ float lds_s2 [2][32 * GH];
  __shared__ float lds_ee [50 * GH];

  const int blk  = blockIdx.x;          // b*32 + itile
  const int b    = blk >> 5;
  const int i0   = (blk & 31) << 4;
  const int t    = threadIdx.x;
  const int h    = t >> 5;              // wave == head
  const int lane = t & 31;
  const int iloc = lane & 15;
  const int half = lane >> 4;
  const int rsel = half << 3;

  const int* __restrict__ adj_b = adj + (size_t)b * GN * GN;
  const int* __restrict__ et_b  = et  + (size_t)b * GN * GN;
  const float* __restrict__ s2_b = s2 + (size_t)b * GN * GH;
  const _Float16* __restrict__ whb = wh16T + (size_t)(b * GH + h) * GD * GN;

  // Prologue: DMA tile 0 into buffer 0 while everyone loads edge_emb.
  if (t == 0) {
    tdm_load_2d(lds_off(lds_adj[0]), adj_b + (size_t)i0 * GN, 32, 16, GN, GN, GN, 2);
    tdm_load_2d(lds_off(lds_et[0]),  et_b  + (size_t)i0 * GN, 32, 16, GN, GN, GN, 2);
    tdm_load_2d(lds_off(lds_s2[0]),  s2_b,  32 * GH, 1, 32 * GH, 1, 32 * GH, 2);
  }
  for (int idx = t; idx < 50 * GH; idx += 256) lds_ee[idx] = edge_emb[idx];

  const float s1i = s1[(b * GN + i0 + iloc) * GH + h];
  float m_run = -1e30f, l_run = 0.f;
  v8f acc0 = {}, acc1 = {};

  for (int it = 0; it < GN / 32; ++it) {
    const int jt  = it << 5;
    const int cur = it & 1;
    __syncthreads();                    // previous reads of buf cur^1 complete
    if (t == 0) {
      if (it + 1 < GN / 32) {
        const int jn = jt + 32;
        tdm_load_2d(lds_off(lds_adj[cur ^ 1]), adj_b + (size_t)i0 * GN + jn, 32, 16, GN, GN, GN, 2);
        tdm_load_2d(lds_off(lds_et[cur ^ 1]),  et_b  + (size_t)i0 * GN + jn, 32, 16, GN, GN, GN, 2);
        tdm_load_2d(lds_off(lds_s2[cur ^ 1]),  s2_b + (size_t)jn * GH, 32 * GH, 1, 32 * GH, 1, 32 * GH, 2);
        __builtin_amdgcn_s_wait_tensorcnt((short)3);   // tile `it` landed
      } else {
        __builtin_amdgcn_s_wait_tensorcnt((short)0);   // final tile landed
      }
    }
    __syncthreads();

    // ---- scores for this lane's 16 (i, j) pairs (f16 A-operand K order) ----
    float sc[16];
    float tmax = -1e30f;
#pragma unroll
    for (int kk = 0; kk < 16; ++kk) {
      const int jc = (half << 4) + kk;                 // 0..31 within tile
      float e = s1i + lds_s2[cur][jc * GH + h]
                    + lds_ee[lds_et[cur][iloc * 32 + jc] * GH + h];
      e = (e > 0.f) ? e : 0.2f * e;                    // leaky_relu(0.2)
      e = (lds_adj[cur][iloc * 32 + jc] == 0) ? -1e9f : e;
      sc[kk] = e;
      tmax = fmaxf(tmax, e);
    }
    tmax = fmaxf(tmax, __shfl_xor(tmax, 16, 32));      // merge lane pair (row)
    const float m_new = fmaxf(m_run, tmax);
    const float scale = __expf(m_run - m_new);

    // rescale accumulators per-row (row of VGPR r = r + rsel)
#pragma unroll
    for (int r = 0; r < 8; ++r) {
      const float sr = __shfl(scale, r + rsel, 32);
      acc0[r] *= sr;
      acc1[r] *= sr;
    }

    float rowsum = 0.f;
    v16h a;
#pragma unroll
    for (int kk = 0; kk < 16; ++kk) {
      const float p = __expf(sc[kk] - m_new);
      rowsum += p;
      a[kk] = (_Float16)p;
    }
    rowsum += __shfl_xor(rowsum, 16, 32);
    l_run = l_run * scale + rowsum;
    m_run = m_new;

    // ---- B operands: WhT[b,h,d, jt+half*16 .. +15], contiguous 32B ----
    const _Float16* bp = whb + (size_t)iloc * GN + jt + half * 16;
    const v16h b0 = *(const v16h*)bp;
    const v16h b1 = *(const v16h*)(bp + (size_t)16 * GN);
    acc0 = __builtin_amdgcn_wmma_f32_16x16x32_f16(false, a, false, b0, (short)0, acc0, false, false);
    acc1 = __builtin_amdgcn_wmma_f32_16x16x32_f16(false, a, false, b1, (short)0, acc1, false, false);
  }

  const float inv = 1.f / l_run;   // all-masked rows -> uniform, matches ref
#pragma unroll
  for (int r = 0; r < 8; ++r) {
    const float ir = __shfl(inv, r + rsel, 32);
    const int row  = i0 + r + rsel;
    float* op = out + (size_t)(b * GN + row) * GF + h * GD + iloc;
    op[0]  = acc0[r] * ir;
    op[16] = acc1[r] * ir;
  }
}

// ---------------------------------------------------------------------------
// Kernel 4: in-place LayerNorm(256) + ELU. One block per node.
// ---------------------------------------------------------------------------
__global__ __launch_bounds__(256)
void gat_ln_elu(float* __restrict__ out, const float* __restrict__ gamma,
                const float* __restrict__ beta) {
  __shared__ float rs[8], rq[8];
  const int row = blockIdx.x;
  const int t = threadIdx.x;
  const float v = out[(size_t)row * GF + t];
  float s = v, q = v * v;
#pragma unroll
  for (int o = 16; o > 0; o >>= 1) {
    s += __shfl_xor(s, o, 32);
    q += __shfl_xor(q, o, 32);
  }
  const int wave = t >> 5, lane = t & 31;
  if (lane == 0) { rs[wave] = s; rq[wave] = q; }
  __syncthreads();
  if (t == 0) {
    float a = 0.f, c = 0.f;
    for (int w = 0; w < 8; ++w) { a += rs[w]; c += rq[w]; }
    rs[0] = a; rq[0] = c;
  }
  __syncthreads();
  const float mean = rs[0] * (1.f / 256.f);
  const float var  = rq[0] * (1.f / 256.f) - mean * mean;
  float y = (v - mean) * rsqrtf(var + 1e-5f) * gamma[t] + beta[t];
  y = (y > 0.f) ? y : (__expf(y) - 1.f);   // ELU(alpha=1)
  out[(size_t)row * GF + t] = y;
}

// ---------------------------------------------------------------------------
extern "C" void kernel_launch(void* const* d_in, const int* in_sizes, int n_in,
                              void* d_out, int out_size, void* d_ws, size_t ws_size,
                              hipStream_t stream) {
  (void)in_sizes; (void)n_in; (void)out_size; (void)ws_size;
  const float* x     = (const float*)d_in[0];
  const int*   adj   = (const int*)d_in[1];
  const int*   et    = (const int*)d_in[2];
  const float* W     = (const float*)d_in[3];
  const float* a1    = (const float*)d_in[4];
  const float* a2    = (const float*)d_in[5];
  const float* ee    = (const float*)d_in[6];
  const float* gamma = (const float*)d_in[7];
  const float* beta  = (const float*)d_in[8];
  float* out = (float*)d_out;

  char* ws = (char*)d_ws;
  const size_t whElems = (size_t)GB * GN * GF;        // 2M
  float*    wh32  = (float*)ws;                        // 8.39 MB
  _Float16* wh16T = (_Float16*)(ws + whElems * 4);     // 4.19 MB
  float*    s1    = (float*)(ws + whElems * 4 + whElems * 2);
  float*    s2    = s1 + (size_t)GB * GN * GH;

  gat_gemm_wh<<<(GB * GN) / 16, 256, 0, stream>>>(x, W, wh32, wh16T);
  gat_scores <<<GB * GN,        256, 0, stream>>>(wh32, a1, a2, s1, s2);
  gat_attn   <<<GB * (GN / 16), 256, 0, stream>>>(s1, s2, wh16T, adj, et, ee, out);
  gat_ln_elu <<<GB * GN,        256, 0, stream>>>(out, gamma, beta);
}
